// BahdanauAttention_23304492548713
// MI455X (gfx1250) — matrix operand order repaired and checked by reading, not compiled, not run on hardware
//
#include <hip/hip_runtime.h>
#include <math.h>

// ---------------------------------------------------------------------------
// Bahdanau attention, MI455X (gfx1250, wave32, WMMA + double-buffered TDM).
// Core GEMM: v[b,s,h] = value[b,s,:] @ Wv[:,h]  (65536 x 2048 x 1024)
// bf16 inputs / f32 accum via v_wmma_f32_16x16x32_bf16; A tiles streamed into
// LDS by the Tensor Data Mover (tensor_load_to_lds) one stage ahead of the
// compute, synchronized with TENSORcnt (s_wait_tensorcnt 1/0).
// ---------------------------------------------------------------------------

typedef __attribute__((ext_vector_type(16))) __bf16 v16bf;
typedef __attribute__((ext_vector_type(8)))  __bf16 bf16x8;
typedef __attribute__((ext_vector_type(8)))  float  v8f;
typedef __attribute__((ext_vector_type(4)))  float  f32x4;
typedef __attribute__((ext_vector_type(4)))  unsigned int u32x4;
typedef __attribute__((ext_vector_type(8)))  int    i32x8;
typedef __attribute__((ext_vector_type(4)))  int    i32x4;

#define HD    1024      // H
#define TWOH  2048      // 2H
#define BB    32        // batch
#define SS    2048      // sequence

// LDS A tile: 32 rows x 64 f32, padded by TDM to 68-dword row stride
// (pad_interval = 64 dwords, pad_amount = 4 dwords) -> conflict-free b128
// reads across the 16 lanes of a half-wave and 16-byte aligned rows.
#define AROW   68
#define ABUF   (32 * AROW)    // dwords per buffer (8704 B, 16B-aligned)

// ---------------------------------------------------------------------------
// Kernel 1: q = query @ Wq   ([32,2048] @ [2048,1024] -> [32,1024])
// ---------------------------------------------------------------------------
__global__ __launch_bounds__(256)
void prep_q_kernel(const float* __restrict__ query,
                   const float* __restrict__ Wq,
                   float* __restrict__ qout) {
    const int b = blockIdx.y;
    const int h = blockIdx.x * 256 + threadIdx.x;
    const float* qrow = query + (size_t)b * TWOH;
    float acc = 0.f;
    for (int k = 0; k < TWOH; ++k)
        acc = fmaf(qrow[k], Wq[(size_t)k * HD + h], acc);
    qout[(size_t)b * HD + h] = acc;
}

// ---------------------------------------------------------------------------
// Kernel 2: WvT[n][k] = (bf16) Wv[k][n]   (transposed + converted, 4 MiB, L2-hot)
// ---------------------------------------------------------------------------
__global__ __launch_bounds__(256)
void prep_wvt_kernel(const float* __restrict__ Wv,
                     __bf16* __restrict__ WvT) {
    const int idx = blockIdx.x * 256 + threadIdx.x;   // over HD*TWOH
    const int n = idx / TWOH;
    const int k = idx % TWOH;
    WvT[idx] = (__bf16)Wv[(size_t)k * HD + n];
}

// ---------------------------------------------------------------------------
// Kernel 3: main fused GEMM + score kernel.
// Workgroup: 256 threads (8 wave32). Tile: 32 s-rows x 1024 h-cols, K=2048.
// Per wave: 2 M-tiles x 8 N-tiles (wave w owns h in [w*128, w*128+128)).
// ---------------------------------------------------------------------------
__global__ __launch_bounds__(256)
void bahdanau_gemm_score(const float* __restrict__ value,          // [B,S,2H]
                         const unsigned char* __restrict__ padmask,// [B,S] bool
                         const float* __restrict__ pre_cov,        // [B,S]
                         const __bf16* __restrict__ WvT,           // [H][2H] bf16
                         const float* __restrict__ qproj,          // [B,H]
                         const float* __restrict__ We,             // [H]
                         const float* __restrict__ Wc,             // [H]
                         __bf16* __restrict__ vout,                // [B,S,H] bf16
                         float* __restrict__ scores)               // [B,S]
{
    __shared__ float ldsAf[2 * ABUF + 8];   // double-buffered A tile (f32)
    __shared__ float sAcc[32];

    const int b    = blockIdx.y;
    const int s0   = blockIdx.x * 32;
    const int tid  = threadIdx.x;
    const int wave = tid >> 5;
    const int lane = tid & 31;
    const int nl   = lane & 15;   // N within a tile / M-row for A frag
    const int half = lane >> 4;   // lane group
    const int wbase = wave * 128; // this wave's h base

    if (tid < 32) sAcc[tid] = 0.f;

    v8f acc[2][8];
    #pragma unroll
    for (int mt = 0; mt < 2; ++mt)
        #pragma unroll
        for (int nt = 0; nt < 8; ++nt)
            acc[mt][nt] = v8f{};

#if __has_builtin(__builtin_amdgcn_tensor_load_to_lds)
    // ---- Tensor Data Mover descriptor (constant parts) ----
    // Group 0: count=1 | lds_addr | global_addr(57b) | type=2
    // Group 1: data_size=4B, pad_enable, pad_interval=64dw, pad_amount=4dw,
    //          tensor_dim = 64 x 32, tile_dim = 64 x 32, dim0 stride = 2048.
    const unsigned int ldsBase = (unsigned int)(uintptr_t)&ldsAf[0];
    i32x8 g1;
    g1[0] = (2 << 16) | (1 << 20) | (5 << 22) | (3 << 25);
    g1[1] = (64 & 0xffff) << 16;          // tensor_dim0[15:0]
    g1[2] = (32 << 16);                   // tensor_dim0[31:16]=0 | tensor_dim1[15:0]
    g1[3] = (64 << 16);                   // tensor_dim1[31:16]=0 | tile_dim0
    g1[4] = 32;                           // tile_dim1 | tile_dim2=0
    g1[5] = 2048;                         // tensor_dim0_stride[31:0] (elements)
    g1[6] = 0;
    g1[7] = 0;
    const i32x4 gz4 = i32x4{};
#if defined(__clang_major__) && (__clang_major__ >= 23)
    const i32x8 gz8 = i32x8{};
#endif
    // Issue one 2D-tile DMA: value[b, s0:s0+32, k0:k0+64] -> LDS buffer `buf`.
    auto issueTdm = [&](int k0, int buf) {
        const unsigned long long gaddr = (unsigned long long)(uintptr_t)
            (value + ((size_t)b * SS + s0) * TWOH + k0);
        u32x4 g0;
        g0[0] = 1u;                                       // count=1 (user mode)
        g0[1] = ldsBase + (unsigned int)(buf * ABUF * 4); // D#.lds_addr
        g0[2] = (unsigned int)gaddr;                      // global_addr[31:0]
        g0[3] = (unsigned int)((gaddr >> 32) & 0x1FFFFFFull) | (2u << 30);
#if defined(__clang_major__) && (__clang_major__ >= 23)
        __builtin_amdgcn_tensor_load_to_lds(g0, g1, gz4, gz4, gz8, 0);
#else
        __builtin_amdgcn_tensor_load_to_lds(g0, g1, gz4, gz4, 0);
#endif
    };
    if (wave == 0) issueTdm(0, 0);        // prologue: prefetch stage 0
#else
    // Fallback cooperative-load mapping: thread -> (row 0..31, 8 contiguous K)
    const int rowA = tid >> 3;
    const int colA = (tid & 7) * 8;
    const float* aSrc = value + ((size_t)b * SS + s0 + rowA) * TWOH + colA;
#endif

    for (int k0 = 0; k0 < TWOH; k0 += 64) {
        const int cur = (k0 >> 6) & 1;

#if __has_builtin(__builtin_amdgcn_tensor_load_to_lds)
        if (wave == 0) {
            if (k0 + 64 < TWOH) {
                issueTdm(k0 + 64, cur ^ 1);          // prefetch next stage
                __builtin_amdgcn_s_wait_tensorcnt(1); // current buffer landed
            } else {
                __builtin_amdgcn_s_wait_tensorcnt(0); // drain on last stage
            }
        }
#else
        f32x4 f0 = *(const f32x4*)(aSrc + k0);
        f32x4 f1 = *(const f32x4*)(aSrc + k0 + 4);
        *(f32x4*)&ldsAf[cur * ABUF + rowA * AROW + colA]     = f0;
        *(f32x4*)&ldsAf[cur * ABUF + rowA * AROW + colA + 4] = f1;
#endif
        __syncthreads();   // stage `cur` visible to all waves

        const float* aBuf = &ldsAf[cur * ABUF];
        #pragma unroll
        for (int kk = 0; kk < 64; kk += 32) {
            // A fragments (ISA 16-bit A 16x32 layout):
            //   elems 0..7  <-> K = kk + half*8 + e
            //   elems 8..15 <-> K = kk + 16 + half*8 + (e-8)
            v16bf afrag[2];
            #pragma unroll
            for (int mt = 0; mt < 2; ++mt) {
                const float* ap = &aBuf[(mt * 16 + nl) * AROW + kk + half * 8];
                f32x4 a0 = *(const f32x4*)ap;
                f32x4 a1 = *(const f32x4*)(ap + 4);
                f32x4 a2 = *(const f32x4*)(ap + 16);
                f32x4 a3 = *(const f32x4*)(ap + 20);
                v16bf af;
                af[0]  = (__bf16)a0[0]; af[1]  = (__bf16)a0[1];
                af[2]  = (__bf16)a0[2]; af[3]  = (__bf16)a0[3];
                af[4]  = (__bf16)a1[0]; af[5]  = (__bf16)a1[1];
                af[6]  = (__bf16)a1[2]; af[7]  = (__bf16)a1[3];
                af[8]  = (__bf16)a2[0]; af[9]  = (__bf16)a2[1];
                af[10] = (__bf16)a2[2]; af[11] = (__bf16)a2[3];
                af[12] = (__bf16)a3[0]; af[13] = (__bf16)a3[1];
                af[14] = (__bf16)a3[2]; af[15] = (__bf16)a3[3];
                afrag[mt] = af;
            }
            #pragma unroll
            for (int nt = 0; nt < 8; ++nt) {
                // B fragment straight from L2-hot WvT (K contiguous per lane):
                //   lane: column n = wbase+nt*16+nl, elems e<8 <-> K = kk + half*8 + e, e>=8 <-> K = kk + 16 + half*8 + (e-8)
                const __bf16* bp = WvT
                    + (size_t)(wbase + nt * 16 + nl) * TWOH + k0 + kk + half * 8;
                bf16x8 blo = *(const bf16x8*)bp;
                bf16x8 bhi = *(const bf16x8*)(bp + 16);
                v16bf bfrag = __builtin_shufflevector(blo, bhi,
                    0,1,2,3,4,5,6,7,8,9,10,11,12,13,14,15);
                acc[0][nt] = __builtin_amdgcn_wmma_f32_16x16x32_bf16(
                    false, afrag[0], false, bfrag, (short)0, acc[0][nt], false, false);
                acc[1][nt] = __builtin_amdgcn_wmma_f32_16x16x32_bf16(
                    false, afrag[1], false, bfrag, (short)0, acc[1][nt], false, false);
            }
        }
        __syncthreads();   // buffer `cur` retired; next iter may DMA into cur^1
    }

    // ---------------- epilogue ----------------
    // C layout: element r of v8f, lane l: M = r + 8*half, N = nl.
    float qn[8], wev[8], wcv[8];
    #pragma unroll
    for (int nt = 0; nt < 8; ++nt) {
        const int n = wbase + nt * 16 + nl;
        qn[nt]  = qproj[(size_t)b * HD + n];
        wev[nt] = We[n];
        wcv[nt] = Wc[n];
    }
    float covm[2][8];
    #pragma unroll
    for (int mt = 0; mt < 2; ++mt)
        #pragma unroll
        for (int r = 0; r < 8; ++r)
            covm[mt][r] = pre_cov[(size_t)b * SS + s0 + mt * 16 + r + half * 8];

    float partial[2][8];
    #pragma unroll
    for (int mt = 0; mt < 2; ++mt)
        #pragma unroll
        for (int r = 0; r < 8; ++r)
            partial[mt][r] = 0.f;

    #pragma unroll
    for (int mt = 0; mt < 2; ++mt) {
        #pragma unroll
        for (int nt = 0; nt < 8; ++nt) {
            const int n = wbase + nt * 16 + nl;
            v8f c = acc[mt][nt];
            #pragma unroll
            for (int r = 0; r < 8; ++r) {
                const int srow = s0 + mt * 16 + r + half * 8;
                const float vv = c[r];
                vout[((size_t)b * SS + srow) * HD + n] = (__bf16)vv; // projected value
                const float feat = vv + qn[nt] + covm[mt][r] * wcv[nt];
                partial[mt][r] += tanhf(feat) * wev[nt];
            }
        }
    }

    // Reduce over the 16 lanes of each half (they hold distinct n).
    #pragma unroll
    for (int mt = 0; mt < 2; ++mt)
        #pragma unroll
        for (int r = 0; r < 8; ++r) {
            float p = partial[mt][r];
            p += __shfl_xor(p, 1, 32);
            p += __shfl_xor(p, 2, 32);
            p += __shfl_xor(p, 4, 32);
            p += __shfl_xor(p, 8, 32);
            partial[mt][r] = p;
        }
    if (nl == 0) {  // lanes 0 and 16 of each wave carry half-specific m rows
        #pragma unroll
        for (int mt = 0; mt < 2; ++mt)
            #pragma unroll
            for (int r = 0; r < 8; ++r)
                atomicAdd(&sAcc[mt * 16 + r + half * 8], partial[mt][r]);
    }
    __syncthreads();
    if (tid < 32) {
        const int s = s0 + tid;
        float sc = sAcc[tid];
        if (padmask[(size_t)b * SS + s]) sc = -INFINITY;
        scores[(size_t)b * SS + s] = sc;
    }
}

// ---------------------------------------------------------------------------
// Kernel 4: softmax over S per batch + coverage update.
// ---------------------------------------------------------------------------
__global__ __launch_bounds__(256)
void softmax_cov_kernel(const float* __restrict__ scores,
                        const float* __restrict__ pre_cov,
                        float* __restrict__ alphas_ws,
                        float* __restrict__ out_alphas,
                        float* __restrict__ out_newcov) {
    const int b = blockIdx.x;
    const int tid = threadIdx.x;
    __shared__ float red[256];

    float vals[8];
    float lmax = -INFINITY;
    #pragma unroll
    for (int i = 0; i < 8; ++i) {
        vals[i] = scores[(size_t)b * SS + tid + i * 256];
        lmax = fmaxf(lmax, vals[i]);
    }
    red[tid] = lmax; __syncthreads();
    for (int o = 128; o > 0; o >>= 1) {
        if (tid < o) red[tid] = fmaxf(red[tid], red[tid + o]);
        __syncthreads();
    }
    const float m = red[0]; __syncthreads();

    float lsum = 0.f;
    #pragma unroll
    for (int i = 0; i < 8; ++i) {
        vals[i] = __expf(vals[i] - m);
        lsum += vals[i];
    }
    red[tid] = lsum; __syncthreads();
    for (int o = 128; o > 0; o >>= 1) {
        if (tid < o) red[tid] += red[tid + o];
        __syncthreads();
    }
    const float inv = 1.f / red[0];
    #pragma unroll
    for (int i = 0; i < 8; ++i) {
        const int s = tid + i * 256;
        const float a = vals[i] * inv;
        alphas_ws[(size_t)b * SS + s]  = a;
        out_alphas[(size_t)b * SS + s] = a;
        out_newcov[(size_t)b * SS + s] = pre_cov[(size_t)b * SS + s] + a;
    }
}

// ---------------------------------------------------------------------------
// Kernel 5: context[b,h] = sum_s alphas[b,s] * v[b,s,h]
// ---------------------------------------------------------------------------
__global__ __launch_bounds__(256)
void context_kernel(const float* __restrict__ alphas,
                    const __bf16* __restrict__ v,
                    float* __restrict__ out_ctx) {
    const int b = blockIdx.y;
    const int h = blockIdx.x * 256 + threadIdx.x;
    const __bf16* vp = v + (size_t)b * SS * HD + h;
    const float*  ap = alphas + (size_t)b * SS;
    float acc = 0.f;
    for (int s = 0; s < SS; ++s)
        acc = fmaf(ap[s], (float)vp[(size_t)s * HD], acc);
    out_ctx[(size_t)b * HD + h] = acc;
}

// ---------------------------------------------------------------------------
extern "C" void kernel_launch(void* const* d_in, const int* in_sizes, int n_in,
                              void* d_out, int out_size, void* d_ws, size_t ws_size,
                              hipStream_t stream) {
    const float*         value   = (const float*)d_in[0];         // [B,S,2H]
    const float*         query   = (const float*)d_in[1];         // [B,2H]
    const unsigned char* padmask = (const unsigned char*)d_in[2]; // [B,S] bool
    const float*         pre_cov = (const float*)d_in[3];         // [B,S]
    const float*         Wv      = (const float*)d_in[4];         // [2H,H]
    const float*         Wq      = (const float*)d_in[5];         // [2H,H]
    const float*         We      = (const float*)d_in[6];         // [H]
    const float*         Wc      = (const float*)d_in[7];         // [H]

    float* out        = (float*)d_out;
    float* out_ctx    = out;                       // [B,1,H]  = 32768 f32
    float* out_alphas = out + BB * HD;             // [B,S]    = 65536 f32
    float* out_newcov = out_alphas + BB * SS;      // [B,S]    = 65536 f32

    char* p = (char*)d_ws;
    float*  qws    = (float*)p;  p += (size_t)BB * HD * sizeof(float);     // 128 KiB
    __bf16* WvT    = (__bf16*)p; p += (size_t)HD * TWOH * sizeof(__bf16);  // 4 MiB
    float*  scores = (float*)p;  p += (size_t)BB * SS * sizeof(float);     // 256 KiB
    float*  alphas = (float*)p;  p += (size_t)BB * SS * sizeof(float);     // 256 KiB
    __bf16* vws    = (__bf16*)p;                                           // 128 MiB

    prep_q_kernel  <<<dim3(HD / 256, BB), 256, 0, stream>>>(query, Wq, qws);
    prep_wvt_kernel<<<(HD * TWOH) / 256,  256, 0, stream>>>(Wv, WvT);
    bahdanau_gemm_score<<<dim3(SS / 32, BB), 256, 0, stream>>>(
        value, padmask, pre_cov, WvT, qws, We, Wc, vws, scores);
    softmax_cov_kernel<<<BB, 256, 0, stream>>>(scores, pre_cov, alphas,
                                               out_alphas, out_newcov);
    context_kernel<<<dim3(HD / 256, BB), 256, 0, stream>>>(alphas, vws, out_ctx);
}
